// GeneralizedXNorAttention_68556267978966
// MI455X (gfx1250) — compile-verified, matrix operand order checked by reading, not run on hardware
//
#include <hip/hip_runtime.h>
#include <hip/hip_bf16.h>
#include <cstdint>

// ---------------------------------------------------------------------------
// cosFormer linear attention for MI455X (gfx1250, wave32, WMMA).
// All GEMMs use v_wmma_f32_16x16x32_bf16 (bf16 operands, f32 accumulate).
// Inputs/weights pre-converted to bf16 so GEMM inner loops are pure
// async-LDS staging + ds_load_b128 fragment builds + WMMA (no cvt VALU).
// ---------------------------------------------------------------------------

typedef __attribute__((ext_vector_type(16))) __bf16   v16bf;
typedef __attribute__((ext_vector_type(2)))  __bf16   v2bf;
typedef __attribute__((ext_vector_type(8)))  float    v8f;
typedef __attribute__((ext_vector_type(4)))  unsigned u32x4;
typedef __attribute__((ext_vector_type(4)))  float    f32x4;
typedef int b128v __attribute__((vector_size(16)));   // matches async builtin param

union Frag {
    v16bf v;
    unsigned u[8];
    u32x4 q[2];
};

__device__ inline v8f vzero() {
    v8f z;
#pragma unroll
    for (int i = 0; i < 8; ++i) z[i] = 0.0f;
    return z;
}

__device__ inline v8f wmma_bf16(v16bf a, v16bf b, v8f c) {
    // D = A(16x32) * B(32x16) + C, f32 accumulate
    return __builtin_amdgcn_wmma_f32_16x16x32_bf16(false, a, false, b,
                                                   (short)0, c, false, false);
}

// float pair -> packed bf16x2 (prefer hardware v_cvt_pk_bf16_f32)
__device__ inline unsigned pk_bf16(float a, float b) {
#if __has_builtin(__builtin_amdgcn_cvt_pk_bf16_f32)
    v2bf r = __builtin_amdgcn_cvt_pk_bf16_f32(a, b);
    return __builtin_bit_cast(unsigned, r);
#else
    unsigned ua = __builtin_bit_cast(unsigned, a);
    unsigned ub = __builtin_bit_cast(unsigned, b);
    ua = (ua + 0x7FFFu + ((ua >> 16) & 1u)) >> 16;
    ub = (ub + 0x7FFFu + ((ub >> 16) & 1u)) >> 16;
    return (ua & 0xFFFFu) | (ub << 16);
#endif
}
__device__ inline unsigned short f2bf(float a) {
    return (unsigned short)(pk_bf16(a, 0.0f) & 0xFFFFu);
}

// ---------------------------------------------------------------------------
// Async global->LDS copy (CDNA5 GLOBAL_LOAD_ASYNC_TO_LDS_B128), guarded.
// ---------------------------------------------------------------------------
#if __has_builtin(__builtin_amdgcn_global_load_async_to_lds_b128) && \
    __has_builtin(__builtin_amdgcn_s_wait_asynccnt)
#define HAS_ASYNC 1
#endif

__device__ inline void cp16(void* lds, const void* g) {
#ifdef HAS_ASYNC
    __builtin_amdgcn_global_load_async_to_lds_b128(
        (__attribute__((address_space(1))) b128v*)g,
        (__attribute__((address_space(3))) b128v*)lds, 0, 0);
#else
    *(f32x4*)lds = *(const f32x4*)g;
#endif
}

// bf16 fragment from bf16 LDS row slice; p -> [row][ks], row stride 32 elems.
__device__ inline void load_frag(Frag& f, const unsigned short* p) {
    f.q[0] = *(const u32x4*)p;          // K = ks .. ks+7
    f.q[1] = *(const u32x4*)(p + 16);   // K = ks+16 .. ks+23
}

// ---------------------------------------------------------------------------
// Kernel 0: streaming f32 -> bf16 conversion (8 elems/thread)
// ---------------------------------------------------------------------------
__global__ __launch_bounds__(256) void cvt_bf16(const float* __restrict__ src,
                                                unsigned short* __restrict__ dst,
                                                int n8) {
    const int i = blockIdx.x * 256 + threadIdx.x;
    if (i >= n8) return;
    const f32x4* s = (const f32x4*)(src + (size_t)i * 8);
    const f32x4 a = s[0], b = s[1];
    u32x4 o;
    o.x = pk_bf16(a.x, a.y);
    o.y = pk_bf16(a.z, a.w);
    o.z = pk_bf16(b.x, b.y);
    o.w = pk_bf16(b.z, b.w);
    *(u32x4*)(dst + (size_t)i * 8) = o;
}

// ---------------------------------------------------------------------------
// Kernel 1: C(M=32768,512) = A(M,512) @ W(512,512)^T + bias
// A, W bf16; OUT_BF16 selects f32 or bf16 output.
// WG tile 128x128, 8 waves of 64(M)x32(N), K step 32, double-buffered async.
// ---------------------------------------------------------------------------
__device__ inline void stage_tile(const unsigned short* __restrict__ A,
                                  const unsigned short* __restrict__ W,
                                  unsigned short* At, unsigned short* Wt,
                                  int tid, int m0, int n0, int k0) {
#pragma unroll
    for (int i = 0; i < 2; ++i) {
        const int c = tid * 2 + i;       // 0..511
        const int row = c >> 2;          // 0..127
        const int part = c & 3;          // 16B = 8 bf16 each
        cp16(At + row * 32 + part * 8, A + (size_t)(m0 + row) * 512 + k0 + part * 8);
        cp16(Wt + row * 32 + part * 8, W + (size_t)(n0 + row) * 512 + k0 + part * 8);
    }
}

template <bool OUT_BF16>
__global__ __launch_bounds__(256) void gemm512(const unsigned short* __restrict__ A,
                                               const unsigned short* __restrict__ W,
                                               const float* __restrict__ bias,
                                               float* __restrict__ Cf,
                                               unsigned short* __restrict__ Cb) {
    __shared__ __align__(16) unsigned short At[2][128 * 32];
    __shared__ __align__(16) unsigned short Wt[2][128 * 32];
    const int tid = threadIdx.x;
    const int lane = tid & 31;
    const int w = tid >> 5;
    const int wm = w >> 2;              // 0..1 -> M offset 64*wm
    const int wn = w & 3;               // 0..3 -> N offset 32*wn
    const int m0 = blockIdx.x * 128;
    const int n0 = blockIdx.y * 128;
    const int lr = lane & 15;
    const int ks = (lane >> 4) * 8;

    v8f acc[4][2];
#pragma unroll
    for (int i = 0; i < 4; ++i)
#pragma unroll
        for (int j = 0; j < 2; ++j) acc[i][j] = vzero();

    stage_tile(A, W, At[0], Wt[0], tid, m0, n0, 0);
    stage_tile(A, W, At[1], Wt[1], tid, m0, n0, 32);

    for (int kstep = 0; kstep < 16; ++kstep) {
        const int buf = kstep & 1;
#ifdef HAS_ASYNC
        if (kstep < 15) __builtin_amdgcn_s_wait_asynccnt(4);
        else            __builtin_amdgcn_s_wait_asynccnt(0);
#endif
        __syncthreads();
        // preload all fragments into distinct regs so ds_loads clause together
        Frag bfr[2], afr[4];
#pragma unroll
        for (int ni = 0; ni < 2; ++ni)
            load_frag(bfr[ni], &Wt[buf][(wn * 32 + ni * 16 + lr) * 32 + ks]);
#pragma unroll
        for (int mi = 0; mi < 4; ++mi)
            load_frag(afr[mi], &At[buf][(wm * 64 + mi * 16 + lr) * 32 + ks]);
#pragma unroll
        for (int mi = 0; mi < 4; ++mi) {
            acc[mi][0] = wmma_bf16(afr[mi].v, bfr[0].v, acc[mi][0]);
            acc[mi][1] = wmma_bf16(afr[mi].v, bfr[1].v, acc[mi][1]);
        }
        __syncthreads();
        if (kstep + 2 < 16)
            stage_tile(A, W, At[buf], Wt[buf], tid, m0, n0, (kstep + 2) * 32);
    }

    // C layout: vgpr j -> M = j + 8*(lane>>4), N = lane&15
#pragma unroll
    for (int mi = 0; mi < 4; ++mi)
#pragma unroll
        for (int ni = 0; ni < 2; ++ni) {
            const int f = n0 + wn * 32 + ni * 16 + lr;
            const float bv = bias[f];
            const int mb = m0 + wm * 64 + mi * 16 + ((lane >> 4) << 3);
#pragma unroll
            for (int j = 0; j < 8; ++j) {
                if (OUT_BF16)
                    Cb[(size_t)(mb + j) * 512 + f] = f2bf(acc[mi][ni][j] + bv);
                else
                    Cf[(size_t)(mb + j) * 512 + f] = acc[mi][ni][j] + bv;
            }
        }
}

// ---------------------------------------------------------------------------
// Kernel 2: q softmax over D=64 per (t,b,l,h) + sin/cos doubling -> bf16
//   out: qt[t][n=b*8+h][l][j], j<64: p*sin(l), j>=64: p*cos(l)
// ---------------------------------------------------------------------------
__global__ __launch_bounds__(256) void softmax_q(const float* __restrict__ Qp,
                                                 unsigned short* __restrict__ qt) {
    const int t = blockIdx.y;
    const int row = blockIdx.x;                 // b*4096 + l
    const int b = row >> 12, l = row & 4095;
    const int h = threadIdx.x >> 5, lane = threadIdx.x & 31;
    const float* src = Qp + ((size_t)t * 32768 + row) * 512 + h * 64 + lane * 2;
    const float x0 = src[0], x1 = src[1];
    float m = fmaxf(x0, x1);
    for (int off = 16; off > 0; off >>= 1) m = fmaxf(m, __shfl_xor(m, off, 32));
    const float e0 = __expf(x0 - m), e1 = __expf(x1 - m);
    float s = e0 + e1;
    for (int off = 16; off > 0; off >>= 1) s += __shfl_xor(s, off, 32);
    const float inv = 1.0f / s;
    const float ang = 1.57079632679f * (float)(l + 1) * (1.0f / 4096.0f);
    const float sl = __sinf(ang), cl = __cosf(ang);
    unsigned short* dst = qt + (((size_t)t * 64 + b * 8 + h) * 4096 + l) * 128;
    *(unsigned*)(dst + 2 * lane)      = pk_bf16(e0 * inv * sl, e1 * inv * sl);
    *(unsigned*)(dst + 64 + 2 * lane) = pk_bf16(e0 * inv * cl, e1 * inv * cl);
}

// ---------------------------------------------------------------------------
// Kernel 3: k softmax over S=4096 per (t,b,f) + sin/cos -> k~^T bf16
//   out: ktT[t][n][j][s] (S contiguous); j<64: sin, j>=64: cos
// ---------------------------------------------------------------------------
__global__ __launch_bounds__(256) void softmax_k(const float* __restrict__ Kp,
                                                 unsigned short* __restrict__ ktT) {
    const int gw = (blockIdx.x * 256 + threadIdx.x) >> 5;   // 0..8191
    const int lane = threadIdx.x & 31;
    const int t = gw >> 12;
    const int b = (gw >> 9) & 7;
    const int f = gw & 511;
    const float* col = Kp + ((size_t)t * 32768 + (size_t)b * 4096) * 512 + f;
    float m = -3.0e38f;
    for (int s = lane; s < 4096; s += 32) m = fmaxf(m, col[(size_t)s * 512]);
    for (int off = 16; off > 0; off >>= 1) m = fmaxf(m, __shfl_xor(m, off, 32));
    float sum = 0.0f;
    for (int s = lane; s < 4096; s += 32) sum += __expf(col[(size_t)s * 512] - m);
    for (int off = 16; off > 0; off >>= 1) sum += __shfl_xor(sum, off, 32);
    const float inv = 1.0f / sum;
    const int h = f >> 6, d = f & 63;
    unsigned short* d0 = ktT + (((size_t)t * 64 + b * 8 + h) * 128 + d) * 4096;
    unsigned short* d1 = d0 + (size_t)64 * 4096;
    for (int s = lane; s < 4096; s += 32) {
        const float p = __expf(col[(size_t)s * 512] - m) * inv;
        const float ang = 1.57079632679f * (float)(s + 1) * (1.0f / 4096.0f);
        d0[s] = f2bf(p * __sinf(ang));
        d1[s] = f2bf(p * __cosf(ang));
    }
}

// ---------------------------------------------------------------------------
// Kernel 4: kv[t,n] (128x64) = k~^T[t,n] (128xS) @ v_used[n] (Sx64)
// v_used[n,s,d] = VprojFlat[(s*64+n)*64+d]  (pure flat reinterpretation)
// Double-buffered LDS v-tile (1 barrier/step) + A-fragment prefetch.
// ---------------------------------------------------------------------------
#define VT_STRIDE 40   // elements; 80B rows keep 16B alignment, conflict-free
__global__ __launch_bounds__(256) void kv_gemm(const unsigned short* __restrict__ ktT,
                                               const unsigned short* __restrict__ Vp,
                                               float* __restrict__ kvb) {
    __shared__ __align__(16) unsigned short vT[2][64 * VT_STRIDE];
    const int tn = blockIdx.x;          // t*64 + n
    const int n = tn & 63;
    const int tid = threadIdx.x, lane = tid & 31, w = tid >> 5;
    const int lr = lane & 15, ks = (lane >> 4) * 8;
    const unsigned short* Ab =
        ktT + (size_t)tn * 128 * 4096 + (size_t)(w * 16 + lr) * 4096;

    v8f acc[4];
#pragma unroll
    for (int i = 0; i < 4; ++i) acc[i] = vzero();

    const int sl = tid >> 3, part = tid & 7;
    const int d0 = part * 8;

    auto stage = [&](int sbuf, int k0) {
        const unsigned short* g = Vp + ((size_t)(k0 + sl) * 64 + n) * 64 + d0;
        u32x4 x = *(const u32x4*)g;                 // 8 bf16
        const unsigned short* px = (const unsigned short*)&x;
        unsigned short* dst = &vT[sbuf][0];
#pragma unroll
        for (int i = 0; i < 8; ++i)
            dst[(d0 + i) * VT_STRIDE + sl] = px[i];
    };

    stage(0, 0);
    Frag afCur;
    {
        const unsigned short* ap = Ab + ks;
        afCur.q[0] = *(const u32x4*)ap;
        afCur.q[1] = *(const u32x4*)(ap + 16);
    }

    int buf = 0;
    for (int k0 = 0; k0 < 4096; k0 += 32) {
        __syncthreads();                 // vT[buf] ready; prior reads of vT[buf^1] done
        Frag afNext = afCur;
        if (k0 + 32 < 4096) {
            stage(buf ^ 1, k0 + 32);     // overlaps with this step's WMMAs
            const unsigned short* ap = Ab + k0 + 32 + ks;
            afNext.q[0] = *(const u32x4*)ap;
            afNext.q[1] = *(const u32x4*)(ap + 16);
        }
        Frag bfr[4];
#pragma unroll
        for (int ni = 0; ni < 4; ++ni) {
            const unsigned short* bp = &vT[buf][(ni * 16 + lr) * VT_STRIDE + ks];
            bfr[ni].q[0] = *(const u32x4*)bp;
            bfr[ni].q[1] = *(const u32x4*)(bp + 16);
        }
#pragma unroll
        for (int ni = 0; ni < 4; ++ni)
            acc[ni] = wmma_bf16(afCur.v, bfr[ni].v, acc[ni]);
        afCur = afNext;
        buf ^= 1;
    }

    const int mrow = w * 16 + ((lane >> 4) << 3);
#pragma unroll
    for (int ni = 0; ni < 4; ++ni) {
        const int d = ni * 16 + lr;
#pragma unroll
        for (int j = 0; j < 8; ++j)
            kvb[(size_t)tn * 8192 + (size_t)(mrow + j) * 64 + d] = acc[ni][j];
    }
}

// ---------------------------------------------------------------------------
// Kernel 5: attn[b,l,h*64+d] = sum_t q~[t,n] (Lx128) @ kv[t,n] (128x64)
// Output written as bf16 (consumed by the final Wout GEMM).
// ---------------------------------------------------------------------------
#define KV_STRIDE 264  // elements; 528B rows (33*16), inner offsets 16B-aligned
__global__ __launch_bounds__(256) void attn_gemm(const unsigned short* __restrict__ qt,
                                                 const float* __restrict__ kvb,
                                                 unsigned short* __restrict__ attn) {
    __shared__ __align__(16) unsigned short kvT[64 * KV_STRIDE];
    const int n = blockIdx.y;
    const int l0 = blockIdx.x * 128;
    const int tid = threadIdx.x, lane = tid & 31, w = tid >> 5;
    const int lr = lane & 15, ks = (lane >> 4) * 8;

    for (int i = tid; i < 16384; i += 256) {
        const int t = i >> 13, j = (i >> 6) & 127, d = i & 63;
        kvT[d * KV_STRIDE + t * 128 + j] =
            f2bf(kvb[(((size_t)t * 64 + n) * 128 + j) * 64 + d]);
    }
    __syncthreads();

    v8f acc[4];
#pragma unroll
    for (int i = 0; i < 4; ++i) acc[i] = vzero();

    for (int t = 0; t < 2; ++t) {
        const unsigned short* Ab =
            qt + (((size_t)t * 64 + n) * 4096 + l0 + w * 16 + lr) * 128;
        for (int kk = 0; kk < 128; kk += 32) {
            Frag af;
            const unsigned short* ap = Ab + kk + ks;
            af.q[0] = *(const u32x4*)ap;
            af.q[1] = *(const u32x4*)(ap + 16);
            Frag bfr[4];
#pragma unroll
            for (int ni = 0; ni < 4; ++ni) {
                const unsigned short* bp =
                    &kvT[(ni * 16 + lr) * KV_STRIDE + t * 128 + kk + ks];
                bfr[ni].q[0] = *(const u32x4*)bp;
                bfr[ni].q[1] = *(const u32x4*)(bp + 16);
            }
#pragma unroll
            for (int ni = 0; ni < 4; ++ni)
                acc[ni] = wmma_bf16(af.v, bfr[ni].v, acc[ni]);
        }
    }

    const int b = n >> 3, h = n & 7;
    const int mrow = w * 16 + ((lane >> 4) << 3);
#pragma unroll
    for (int ni = 0; ni < 4; ++ni) {
        const int e = h * 64 + ni * 16 + lr;
#pragma unroll
        for (int j = 0; j < 8; ++j) {
            const int l = l0 + mrow + j;
            attn[((size_t)b * 4096 + l) * 512 + e] = f2bf(acc[ni][j]);
        }
    }
}

// ---------------------------------------------------------------------------
// Launch. Workspace layout (bytes from d_ws), ~648 MiB:
//   qbf/kbf/vbf bf16 @0/32/64 MiB (32 MiB each; qbf reused as attn later)
//   weights bf16 @96 MiB (Wq 1, Wk 1, Wv 0.5, Wout 0.5)
//   Qproj f32 @100 MiB (128), Kproj f32 @228 MiB (128)
//   Vproj bf16 @356 MiB (32), q~ bf16 @388 MiB (128), k~^T bf16 @516 MiB (128)
//   kv f32 @644 MiB (4)
// ---------------------------------------------------------------------------
extern "C" void kernel_launch(void* const* d_in, const int* in_sizes, int n_in,
                              void* d_out, int out_size, void* d_ws, size_t ws_size,
                              hipStream_t stream) {
    (void)in_sizes; (void)n_in; (void)out_size; (void)ws_size;
    const float* query = (const float*)d_in[0];
    const float* key   = (const float*)d_in[1];
    const float* value = (const float*)d_in[2];
    const float* Wq    = (const float*)d_in[3];
    const float* bq    = (const float*)d_in[4];
    const float* Wk    = (const float*)d_in[5];
    const float* bk    = (const float*)d_in[6];
    const float* Wv    = (const float*)d_in[7];
    const float* bv    = (const float*)d_in[8];
    const float* Wout  = (const float*)d_in[9];
    const float* bout  = (const float*)d_in[10];

    char* ws = (char*)d_ws;
    const size_t MB = 1024ull * 1024ull;
    unsigned short* qbf   = (unsigned short*)(ws);
    unsigned short* kbf   = (unsigned short*)(ws + 32 * MB);
    unsigned short* vbf   = (unsigned short*)(ws + 64 * MB);
    unsigned short* Wqb   = (unsigned short*)(ws + 96 * MB);
    unsigned short* Wkb   = (unsigned short*)(ws + 97 * MB);
    unsigned short* Wvb   = (unsigned short*)(ws + 98 * MB);
    unsigned short* Wob   = (unsigned short*)(ws + 99 * MB);
    float*          Qproj = (float*)(ws + 100 * MB);
    float*          Kproj = (float*)(ws + 228 * MB);
    unsigned short* Vpb   = (unsigned short*)(ws + 356 * MB);
    unsigned short* qtb   = (unsigned short*)(ws + 388 * MB);
    unsigned short* ktTb  = (unsigned short*)(ws + 516 * MB);
    float*          kvb   = (float*)(ws + 644 * MB);
    unsigned short* attnb = qbf;   // aliases dead qbf (32 MiB)

    // --- pre-convert activations & weights to bf16 ---
    const int nAct8 = (8 * 4096 * 512) / 8;        // 2,097,152
    cvt_bf16<<<(nAct8 + 255) / 256, 256, 0, stream>>>(query, qbf, nAct8);
    cvt_bf16<<<(nAct8 + 255) / 256, 256, 0, stream>>>(key,   kbf, nAct8);
    cvt_bf16<<<(nAct8 + 255) / 256, 256, 0, stream>>>(value, vbf, nAct8);
    const int nW8 = (512 * 512) / 8;               // 32,768
    cvt_bf16<<<(2 * nW8 + 255) / 256, 256, 0, stream>>>(Wq, Wqb, 2 * nW8);
    cvt_bf16<<<(2 * nW8 + 255) / 256, 256, 0, stream>>>(Wk, Wkb, 2 * nW8);
    cvt_bf16<<<(nW8 + 255) / 256, 256, 0, stream>>>(Wv, Wvb, nW8);
    cvt_bf16<<<(nW8 + 255) / 256, 256, 0, stream>>>(Wout, Wob, nW8);

    // --- projections ---
    const dim3 gg(256, 4);   // (32768/128, 512/128)
    gemm512<true><<<gg, 256, 0, stream>>>(vbf, Wvb, bv, nullptr, Vpb);
    for (int t = 0; t < 2; ++t) {
        gemm512<false><<<gg, 256, 0, stream>>>(qbf, Wqb + (size_t)t * 512 * 512,
                                               bq + t * 512,
                                               Qproj + (size_t)t * 32768 * 512, nullptr);
        gemm512<false><<<gg, 256, 0, stream>>>(kbf, Wkb + (size_t)t * 512 * 512,
                                               bk + t * 512,
                                               Kproj + (size_t)t * 32768 * 512, nullptr);
    }

    // --- softmaxes + cosFormer reweighting ---
    softmax_q<<<dim3(32768, 2), 256, 0, stream>>>(Qproj, qtb);
    softmax_k<<<1024, 256, 0, stream>>>(Kproj, ktTb);

    // --- linear attention ---
    kv_gemm<<<128, 256, 0, stream>>>(ktTb, Vpb, kvb);
    attn_gemm<<<dim3(32, 64), 256, 0, stream>>>(qtb, kvb, attnb);

    // --- output projection ---
    gemm512<false><<<gg, 256, 0, stream>>>(attnb, Wob, bout, (float*)d_out, nullptr);
}
